// NodeAttentionLayer_39702677684723
// MI455X (gfx1250) — matrix operand order repaired and checked by reading, not compiled, not run on hardware
//
#include <hip/hip_runtime.h>
#include <hip/hip_bf16.h>

typedef __attribute__((ext_vector_type(16))) _Float16 v16h;
typedef __attribute__((ext_vector_type(8)))  _Float16 v8h;
typedef __attribute__((ext_vector_type(8)))  float    v8f;
typedef __attribute__((ext_vector_type(4)))  float    v4f;
typedef __attribute__((ext_vector_type(4)))  int      v4i;

#define N_NODES 8192
#define F_IN    256
#define F_OUT   64
#define ALPHA   0.2f
#define LOG2E   1.44269504088896340736f

// ---------------------------------------------------------------------------
// Kernel 1: h = X @ W (16x64 tile per wave, WMMA f16 -> f32),
//           e2 = (h @ a_slice) * log2(e), and (o-path) h_oT f16.
// grid = (64, 2): y==0 -> t path, y==1 -> o path. block = 256 (8 waves).
// ---------------------------------------------------------------------------
__global__ __launch_bounds__(256) void gat_proj(
    const float* __restrict__ t_in, const float* __restrict__ o_in,
    const float* __restrict__ W_t,  const float* __restrict__ W_o,
    const float* __restrict__ a,
    float* __restrict__ e_t, float* __restrict__ e_o,
    _Float16* __restrict__ h_oT)
{
    const bool  is_o = (blockIdx.y == 1);
    const float* X  = is_o ? o_in : t_in;
    const float* W  = is_o ? W_o  : W_t;
    const float* av = a + (is_o ? F_OUT : 0);
    float*     evec = is_o ? e_o : e_t;

    __shared__ _Float16 WT[F_OUT * F_IN];     // 32 KB: W transposed, f16
    __shared__ float    Htile[8][16 * F_OUT]; // 32 KB: per-wave D tile

    const int tid = threadIdx.x;
    for (int idx = tid; idx < F_IN * F_OUT; idx += 256) {
        int k = idx >> 6, n = idx & 63;
        WT[n * F_IN + k] = (_Float16)W[idx];
    }
    __syncthreads();

    const int lane = tid & 31;
    const int wv   = tid >> 5;
    const int hi   = lane >> 4;
    const int m    = lane & 15;
    const int row0 = (blockIdx.x * 8 + wv) * 16;

    v8f acc[4] = {};
    for (int k0 = 0; k0 < F_IN; k0 += 32) {
        // A fragment: 16x32 f16. Lane groups: K = k0+8*hi+[0,8) and +16
        const float* arow = X + (size_t)(row0 + m) * F_IN + k0 + 8 * hi;
        v4f a0 = *(const v4f*)(arow);
        v4f a1 = *(const v4f*)(arow + 4);
        v4f a2 = *(const v4f*)(arow + 16);
        v4f a3 = *(const v4f*)(arow + 20);
        v16h af;
        #pragma unroll
        for (int i = 0; i < 4; ++i) {
            af[i]      = (_Float16)a0[i];
            af[4 + i]  = (_Float16)a1[i];
            af[8 + i]  = (_Float16)a2[i];
            af[12 + i] = (_Float16)a3[i];
        }
        // B fragments from LDS (two ds_load_b128 each): lane=n, K=k0+8*hi+v (v<8), k0+16+8*hi+(v-8) (v>=8)
        #pragma unroll
        for (int c = 0; c < 4; ++c) {
            const _Float16* bp = &WT[(m + 16 * c) * F_IN + k0 + 8 * hi];
            v8h b0 = *(const v8h*)bp;
            v8h b1 = *(const v8h*)(bp + 16);
            v16h bf;
            #pragma unroll
            for (int i = 0; i < 8; ++i) { bf[i] = b0[i]; bf[8 + i] = b1[i]; }
            acc[c] = __builtin_amdgcn_wmma_f32_16x16x32_f16(
                false, af, false, bf, (short)0, acc[c], false, false);
        }
    }

    // D layout: VGPR r holds row M = r + 8*hi, col = m + 16*c
    float* ht = &Htile[wv][0];
    #pragma unroll
    for (int c = 0; c < 4; ++c)
        #pragma unroll
        for (int r = 0; r < 8; ++r)
            ht[(r + 8 * hi) * F_OUT + m + 16 * c] = acc[c][r];
    __syncthreads();

    // e2[row] = (h[row] . a_slice) * log2(e)   (log2-domain for v_exp_f32)
    if (lane < 16) {
        const float* hr = &Htile[wv][lane * F_OUT];
        float s = 0.f;
        #pragma unroll
        for (int j = 0; j < F_OUT; ++j) s = fmaf(hr[j], av[j], s);
        evec[row0 + lane] = s * LOG2E;
    }
    // o-path: write h_oT[n][k] (K-contiguous f16) for fast B-fragment loads
    if (is_o) {
        for (int idx = lane; idx < 16 * F_OUT; idx += 32) {
            int mm = idx & 15, col = idx >> 4;
            h_oT[(size_t)col * N_NODES + row0 + mm] =
                (_Float16)Htile[wv][mm * F_OUT + col];
        }
    }
}

// ---------------------------------------------------------------------------
// Kernel 1b: global max of e_o2 (softmax upper-bound base). One block.
// ---------------------------------------------------------------------------
__global__ __launch_bounds__(256) void eo_max_kernel(
    const float* __restrict__ e_o2, float* __restrict__ out_max)
{
    __shared__ float red[256];
    const int tid = threadIdx.x;
    float mx = -3.0e38f;
    for (int i = tid; i < N_NODES; i += 256) mx = fmaxf(mx, e_o2[i]);
    red[tid] = mx;
    __syncthreads();
    #pragma unroll
    for (int s = 128; s > 0; s >>= 1) {
        if (tid < s) red[tid] = fmaxf(red[tid], red[tid + s]);
        __syncthreads();
    }
    if (tid == 0) *out_max = red[0];
}

// ---------------------------------------------------------------------------
// Kernel 2: fused masked-softmax attention + aggregation.
// Upper-bound softmax (no online max): M2_row = lrelu(et2 + max(eo2)) bounds
// every unmasked score, so p = exp2(score2 - M2) in (0,1]; masked -> 0.
// One wave per 16-row block; 512 waves. block = 256 (8 waves), grid 64.
// ---------------------------------------------------------------------------
__global__ __launch_bounds__(256) void gat_attn(
    const int*      __restrict__ adj,
    const float*    __restrict__ e_t2,
    const float*    __restrict__ e_o2,
    const _Float16* __restrict__ h_oT,
    const float*    __restrict__ eo_max,
    float*          __restrict__ out)
{
    const int tid  = threadIdx.x;
    const int lane = tid & 31;
    const int wv   = tid >> 5;
    const int hi   = lane >> 4;
    const int m    = lane & 15;
    const int row0 = (blockIdx.x * 8 + wv) * 16;

    const float et2_m = e_t2[row0 + m];
    const float xb    = et2_m + *eo_max;
    const float M2    = fmaxf(xb, ALPHA * xb);   // lrelu upper bound (log2 dom)
    const int*  arow  = adj + (size_t)(row0 + m) * N_NODES;

    v8f   acc[4] = {};
    float lsum   = 0.f;   // local row-sum (lane pair combined at the end)

    for (int k0 = 0; k0 < N_NODES; k0 += 32) {
        const int kb = k0 + 8 * hi;
        // adj is streamed exactly once -> non-temporal; e_o2/h_oT stay cached
        v4i ad0 = __builtin_nontemporal_load((const v4i*)(arow + kb));
        v4i ad1 = __builtin_nontemporal_load((const v4i*)(arow + kb + 4));
        v4i ad2 = __builtin_nontemporal_load((const v4i*)(arow + kb + 16));
        v4i ad3 = __builtin_nontemporal_load((const v4i*)(arow + kb + 20));
        v4f eo0 = *(const v4f*)(e_o2 + kb);
        v4f eo1 = *(const v4f*)(e_o2 + kb + 4);
        v4f eo2 = *(const v4f*)(e_o2 + kb + 16);
        v4f eo3 = *(const v4f*)(e_o2 + kb + 20);
        __builtin_prefetch(arow + kb + 32, 0, 0);   // global_prefetch_b8

        v16h af;
        #pragma unroll
        for (int i = 0; i < 4; ++i) {
            float x0 = et2_m + eo0[i]; x0 = fmaxf(x0, ALPHA * x0);
            float x1 = et2_m + eo1[i]; x1 = fmaxf(x1, ALPHA * x1);
            float x2 = et2_m + eo2[i]; x2 = fmaxf(x2, ALPHA * x2);
            float x3 = et2_m + eo3[i]; x3 = fmaxf(x3, ALPHA * x3);
            float p0 = __builtin_amdgcn_exp2f(x0 - M2);
            float p1 = __builtin_amdgcn_exp2f(x1 - M2);
            float p2 = __builtin_amdgcn_exp2f(x2 - M2);
            float p3 = __builtin_amdgcn_exp2f(x3 - M2);
            p0 = ad0[i] > 0 ? p0 : 0.f;
            p1 = ad1[i] > 0 ? p1 : 0.f;
            p2 = ad2[i] > 0 ? p2 : 0.f;
            p3 = ad3[i] > 0 ? p3 : 0.f;
            lsum += p0 + p1 + p2 + p3;
            af[i]      = (_Float16)p0;   // A-frag: v<8 -> K=8*hi+v
            af[4 + i]  = (_Float16)p1;
            af[8 + i]  = (_Float16)p2;   // v>=8 -> K=16+8*hi+(v-8)
            af[12 + i] = (_Float16)p3;
        }

        // B fragments: h_oT rows are K-contiguous -> two global_load_b128 each
        #pragma unroll
        for (int c = 0; c < 4; ++c) {
            const _Float16* bp = h_oT + (size_t)(m + 16 * c) * N_NODES + k0 + 8 * hi;
            v8h b0 = *(const v8h*)bp;
            v8h b1 = *(const v8h*)(bp + 16);
            v16h bf;
            #pragma unroll
            for (int i = 0; i < 8; ++i) { bf[i] = b0[i]; bf[8 + i] = b1[i]; }
            acc[c] = __builtin_amdgcn_wmma_f32_16x16x32_f16(
                false, af, false, bf, (short)0, acc[c], false, false);
        }
    }

    // combine lane-pair row sums once, normalize, ELU, store
    const float l_row = lsum + __shfl_xor(lsum, 16, 32);
    #pragma unroll
    for (int r = 0; r < 8; ++r) {
        float lr  = __shfl(l_row, 8 * hi + r, 32);
        float inv = 1.0f / fmaxf(lr, 1.0e-37f);
        #pragma unroll
        for (int c = 0; c < 4; ++c) {
            float hv = acc[c][r] * inv;
            float ev = hv > 0.f ? hv : (__expf(hv) - 1.f);
            out[(size_t)(row0 + r + 8 * hi) * F_OUT + m + 16 * c] = ev;
        }
    }
}

// ---------------------------------------------------------------------------
extern "C" void kernel_launch(void* const* d_in, const int* in_sizes, int n_in,
                              void* d_out, int out_size, void* d_ws, size_t ws_size,
                              hipStream_t stream) {
    const float* t_in = (const float*)d_in[0];
    const float* o_in = (const float*)d_in[1];
    const float* W_t  = (const float*)d_in[2];
    const float* W_o  = (const float*)d_in[3];
    const float* a    = (const float*)d_in[4];
    const int*   adj  = (const int*)d_in[5];
    float*       out  = (float*)d_out;

    // workspace: e_t2 (32KB) | e_o2 (32KB) | eo_max (pad 64B) | h_oT f16 (1MB)
    float*    e_t2   = (float*)d_ws;
    float*    e_o2   = e_t2 + N_NODES;
    float*    eo_max = e_o2 + N_NODES;
    _Float16* h_oT   = (_Float16*)(eo_max + 16);

    gat_proj<<<dim3(N_NODES / (16 * 8), 2), 256, 0, stream>>>(
        t_in, o_in, W_t, W_o, a, e_t2, e_o2, h_oT);
    eo_max_kernel<<<dim3(1), 256, 0, stream>>>(e_o2, eo_max);
    gat_attn<<<dim3(N_NODES / (16 * 8)), 256, 0, stream>>>(
        adj, e_t2, e_o2, h_oT, eo_max, out);
}
